// TemporalAttentionLayer_30794915512420
// MI455X (gfx1250) — matrix operand order repaired and checked by reading, not compiled, not run on hardware
//
#include <hip/hip_runtime.h>
#include <hip/hip_bf16.h>

typedef __bf16 bf16;
typedef __attribute__((ext_vector_type(16))) __bf16 v16bf;
typedef __attribute__((ext_vector_type(8)))  float  v8f;

#define WMMA_BF16(a, b, c) \
  __builtin_amdgcn_wmma_f32_16x16x32_bf16(false, (a), false, (b), (short)0, (c), false, false)

constexpr int Bb    = 2;
constexpr int Nseq  = 2048;
constexpr int Cc    = 1024;
constexpr int Hh    = 16;
constexpr int Dd    = 64;
constexpr int Mrows = Bb * Nseq;   // 4096
constexpr int N3    = 3 * Cc;      // 3072

// ---------------------------------------------------------------------------
// Kernel 1: fused QKV projection.  X[4096,1024]f32 @ Wqkv[1024,3072]f32 + b
// -> Q/K/V scattered as bf16 [3][B][H][N][D].
// 128x128 block tile, 8 waves, each wave 32(M) x 64(N) via 2x4 WMMA tiles.
// B tile stored TRANSPOSED in LDS so fragment reads are contiguous b128s.
// ---------------------------------------------------------------------------
__global__ __launch_bounds__(256)
void ta_qkv_gemm(const float* __restrict__ x, const float* __restrict__ w,
                 const float* __restrict__ bias, bf16* __restrict__ qkv) {
  __shared__ bf16 As[128][40];    // 128(M) x 32(K) (+pad)
  __shared__ bf16 BsT[128][40];   // 128(N) x 32(K) transposed (+pad)

  const int tid  = threadIdx.x;
  const int wid  = tid >> 5;
  const int lane = tid & 31;
  const int lh   = lane >> 4;     // lane half (0/1)
  const int ll   = lane & 15;
  const int waveM = wid >> 1;     // 0..3
  const int waveN = wid & 1;      // 0..1
  const int rowBase = blockIdx.y * 128;
  const int colBase = blockIdx.x * 128;

  v8f acc[2][4] = {};

  for (int kk = 0; kk < Cc; kk += 32) {
    __syncthreads();
    // stage A tile 128x32 f32 -> bf16 (row major)
#pragma unroll
    for (int i = 0; i < 4; ++i) {
      int slot = tid + 256 * i;               // 1024 float4 slots
      int r  = slot >> 3;
      int c4 = (slot & 7) << 2;
      float4 t = *(const float4*)(x + (size_t)(rowBase + r) * Cc + kk + c4);
      As[r][c4 + 0] = (bf16)t.x;  As[r][c4 + 1] = (bf16)t.y;
      As[r][c4 + 2] = (bf16)t.z;  As[r][c4 + 3] = (bf16)t.w;
    }
    // stage B tile 32x128 f32 -> bf16, TRANSPOSED: BsT[col][k]
#pragma unroll
    for (int i = 0; i < 4; ++i) {
      int slot = tid + 256 * i;               // 1024 slots x 4 elements
      int col = slot & 127;
      int k0  = (slot >> 7) << 2;             // 0,4,...,28
      alignas(8) bf16 pk[4];
#pragma unroll
      for (int q = 0; q < 4; ++q)
        pk[q] = (bf16)w[(size_t)(kk + k0 + q) * N3 + colBase + col];
      *(float2*)&BsT[col][k0] = *(const float2*)pk;
    }
    __syncthreads();

    // A fragments: 16x32 bf16, lane = row%16, K chunks split by lane half
    v16bf af[2];
#pragma unroll
    for (int mb = 0; mb < 2; ++mb) {
      int row = waveM * 32 + mb * 16 + ll;
#pragma unroll
      for (int j = 0; j < 16; ++j) {
        int k = lh * 16 + j;
        af[mb][j] = As[row][k];
      }
    }
#pragma unroll
    for (int nb = 0; nb < 4; ++nb) {
      v16bf bfrag;
      int col = waveN * 64 + nb * 16 + ll;
#pragma unroll
      for (int j = 0; j < 16; ++j)            // contiguous 32B per lane
        bfrag[j] = BsT[col][lh * 16 + j];
#pragma unroll
      for (int mb = 0; mb < 2; ++mb)
        acc[mb][nb] = WMMA_BF16(af[mb], bfrag, acc[mb][nb]);
    }
  }

  // epilogue: bias + scatter into [3][B][H][N][D] bf16
#pragma unroll
  for (int mb = 0; mb < 2; ++mb) {
#pragma unroll
    for (int nb = 0; nb < 4; ++nb) {
#pragma unroll
      for (int v = 0; v < 8; ++v) {
        int row = rowBase + waveM * 32 + mb * 16 + v + lh * 8;
        int col = colBase + waveN * 64 + nb * 16 + ll;
        float val = acc[mb][nb][v] + bias[col];
        int which = col / Cc;
        int cc    = col % Cc;
        int h = cc / Dd, d = cc % Dd;
        int bidx = row / Nseq, n = row % Nseq;
        size_t idx = ((((size_t)which * Bb + bidx) * Hh + h) * Nseq + n) * Dd + d;
        qkv[idx] = (bf16)val;
      }
    }
  }
}

// ---------------------------------------------------------------------------
// Kernel 2: flash attention.  One block = 128 query rows of one (b,h) head.
// Each wave owns 16 rows; key blocks of 32; online softmax in f32.
// K staged row-major (S-frag reads contiguous); V staged transposed
// (PV-frag reads contiguous).
// ---------------------------------------------------------------------------
__global__ __launch_bounds__(256)
void ta_flash_attn(const bf16* __restrict__ qkv, bf16* __restrict__ attn) {
  __shared__ bf16 Klds[32][72];      // [key][d]
  __shared__ bf16 VldsT[64][40];     // [d][key]  (transposed)
  __shared__ bf16 Plds[8][16][40];   // per-wave P scratch (C-layout -> A-layout)

  const int tid  = threadIdx.x;
  const int wid  = tid >> 5;
  const int lane = tid & 31;
  const int lh   = lane >> 4;
  const int ll   = lane & 15;
  const int bh = blockIdx.y;
  const int b = bh / Hh, h = bh % Hh;

  const size_t headStride = (size_t)Nseq * Dd;
  const bf16* Qp = qkv + ((size_t)(0 * Bb + b) * Hh + h) * headStride;
  const bf16* Kp = qkv + ((size_t)(1 * Bb + b) * Hh + h) * headStride;
  const bf16* Vp = qkv + ((size_t)(2 * Bb + b) * Hh + h) * headStride;

  const int qrow = blockIdx.x * 128 + wid * 16 + ll;

  // load Q fragments (16 rows x 64 deep, two K=32 chunks)
  v16bf aq[2];
#pragma unroll
  for (int c = 0; c < 2; ++c)
#pragma unroll
    for (int j = 0; j < 16; ++j) {
      int k = c * 32 + lh * 16 + j;
      aq[c][j] = Qp[(size_t)qrow * Dd + k];
    }

  v8f o[4] = {};
  float m[8], l[8];
#pragma unroll
  for (int v = 0; v < 8; ++v) { m[v] = -1e30f; l[v] = 0.0f; }

  const float scale = 0.125f;   // 1/sqrt(64)

  for (int kb = 0; kb < Nseq; kb += 32) {
    __syncthreads();
    {
      // K tile 32x64 row-major: one 16B copy per thread
      int r  = tid >> 3;
      int c8 = (tid & 7) << 3;
      *(float4*)&Klds[r][c8] = *(const float4*)(Kp + (size_t)(kb + r) * Dd + c8);
    }
    // V tile 32x64 -> transposed VldsT[d][key], packed 8B stores
#pragma unroll
    for (int i = 0; i < 2; ++i) {
      int slot = tid + 256 * i;            // 512 slots x 4 elements
      int d  = slot & 63;
      int k0 = (slot >> 6) << 2;           // 0,4,...,28
      alignas(8) bf16 pk[4];
#pragma unroll
      for (int q = 0; q < 4; ++q)
        pk[q] = Vp[(size_t)(kb + k0 + q) * Dd + d];
      *(float2*)&VldsT[d][k0] = *(const float2*)pk;
    }
    if (kb + 32 < Nseq) {
      __builtin_prefetch(Kp + (size_t)(kb + 32) * Dd, 0, 1);
      __builtin_prefetch(Vp + (size_t)(kb + 32) * Dd, 0, 1);
    }
    __syncthreads();

    // S = Q * K^T  (16 x 32 scores: two 16x16 WMMA tiles, K-dim 64 = 2 chunks)
    v8f s[2] = {};
#pragma unroll
    for (int n0 = 0; n0 < 2; ++n0) {
#pragma unroll
      for (int kc = 0; kc < 2; ++kc) {
        v16bf bfrag;
#pragma unroll
        for (int j = 0; j < 16; ++j)       // contiguous in j
          bfrag[j] = Klds[n0 * 16 + ll][kc * 32 + lh * 16 + j];
        s[n0] = WMMA_BF16(aq[kc], bfrag, s[n0]);
      }
    }

    // online softmax (row r = v + 8*lh, cols striped across 16 lanes)
#pragma unroll
    for (int v = 0; v < 8; ++v) {
      float s0 = s[0][v] * scale;
      float s1 = s[1][v] * scale;
      float mx = fmaxf(s0, s1);
#pragma unroll
      for (int off = 1; off <= 8; off <<= 1)
        mx = fmaxf(mx, __shfl_xor(mx, off, 32));
      float mnew  = fmaxf(m[v], mx);
      float alpha = __expf(m[v] - mnew);
      m[v] = mnew;
      float p0 = __expf(s0 - mnew);
      float p1 = __expf(s1 - mnew);
      s[0][v] = p0;
      s[1][v] = p1;
      float r = p0 + p1;
#pragma unroll
      for (int off = 1; off <= 8; off <<= 1)
        r += __shfl_xor(r, off, 32);
      l[v] = l[v] * alpha + r;
#pragma unroll
      for (int dt = 0; dt < 4; ++dt)
        o[dt][v] *= alpha;
    }

    // C-layout P -> bf16 A-layout via per-wave LDS scratch
#pragma unroll
    for (int n0 = 0; n0 < 2; ++n0)
#pragma unroll
      for (int v = 0; v < 8; ++v)
        Plds[wid][v + lh * 8][n0 * 16 + ll] = (bf16)s[n0][v];
    asm volatile("s_wait_dscnt 0" ::: "memory");

    v16bf ap;
#pragma unroll
    for (int j = 0; j < 16; ++j) {
      int k = lh * 16 + j;
      ap[j] = Plds[wid][ll][k];
    }

    // O += P * V   (K-dim = 32 keys, 4 output D-tiles)
#pragma unroll
    for (int dt = 0; dt < 4; ++dt) {
      v16bf bfrag;
#pragma unroll
      for (int j = 0; j < 16; ++j)         // contiguous 32B per lane
        bfrag[j] = VldsT[dt * 16 + ll][lh * 16 + j];
      o[dt] = WMMA_BF16(ap, bfrag, o[dt]);
    }
  }

  // normalize and write [B*N, C] bf16 (head-interleaved columns)
#pragma unroll
  for (int v = 0; v < 8; ++v) {
    float inv = 1.0f / l[v];
    int row = blockIdx.x * 128 + wid * 16 + v + lh * 8;
    size_t base = ((size_t)b * Nseq + row) * Cc + (size_t)h * Dd;
#pragma unroll
    for (int dt = 0; dt < 4; ++dt)
      attn[base + dt * 16 + ll] = (bf16)(o[dt][v] * inv);
  }
}

// ---------------------------------------------------------------------------
// Kernel 3: output projection.  attn[4096,1024]bf16 @ Wout[1024,1024]f32 + b
// -> out f32.  Same skeleton as kernel 1, transposed B staging.
// ---------------------------------------------------------------------------
__global__ __launch_bounds__(256)
void ta_out_gemm(const bf16* __restrict__ a, const float* __restrict__ w,
                 const float* __restrict__ bias, float* __restrict__ out) {
  __shared__ bf16 As[128][40];
  __shared__ bf16 BsT[128][40];

  const int tid  = threadIdx.x;
  const int wid  = tid >> 5;
  const int lane = tid & 31;
  const int lh   = lane >> 4;
  const int ll   = lane & 15;
  const int waveM = wid >> 1;
  const int waveN = wid & 1;
  const int rowBase = blockIdx.y * 128;
  const int colBase = blockIdx.x * 128;

  v8f acc[2][4] = {};

  for (int kk = 0; kk < Cc; kk += 32) {
    __syncthreads();
    // stage A tile 128x32 bf16 (already bf16): 512 float4 slots
#pragma unroll
    for (int i = 0; i < 2; ++i) {
      int slot = tid + 256 * i;
      int r  = slot >> 2;
      int c8 = (slot & 3) << 3;
      *(float4*)&As[r][c8] =
          *(const float4*)(a + (size_t)(rowBase + r) * Cc + kk + c8);
    }
    // stage B tile 32x128 f32 -> bf16, TRANSPOSED: BsT[col][k]
#pragma unroll
    for (int i = 0; i < 4; ++i) {
      int slot = tid + 256 * i;
      int col = slot & 127;
      int k0  = (slot >> 7) << 2;
      alignas(8) bf16 pk[4];
#pragma unroll
      for (int q = 0; q < 4; ++q)
        pk[q] = (bf16)w[(size_t)(kk + k0 + q) * Cc + colBase + col];
      *(float2*)&BsT[col][k0] = *(const float2*)pk;
    }
    __syncthreads();

    v16bf af[2];
#pragma unroll
    for (int mb = 0; mb < 2; ++mb) {
      int row = waveM * 32 + mb * 16 + ll;
#pragma unroll
      for (int j = 0; j < 16; ++j) {
        int k = lh * 16 + j;
        af[mb][j] = As[row][k];
      }
    }
#pragma unroll
    for (int nb = 0; nb < 4; ++nb) {
      v16bf bfrag;
      int col = waveN * 64 + nb * 16 + ll;
#pragma unroll
      for (int j = 0; j < 16; ++j)
        bfrag[j] = BsT[col][lh * 16 + j];
#pragma unroll
      for (int mb = 0; mb < 2; ++mb)
        acc[mb][nb] = WMMA_BF16(af[mb], bfrag, acc[mb][nb]);
    }
  }

#pragma unroll
  for (int mb = 0; mb < 2; ++mb) {
#pragma unroll
    for (int nb = 0; nb < 4; ++nb) {
#pragma unroll
      for (int v = 0; v < 8; ++v) {
        int row = rowBase + waveM * 32 + mb * 16 + v + lh * 8;
        int col = colBase + waveN * 64 + nb * 16 + ll;
        out[(size_t)row * Cc + col] = acc[mb][nb][v] + bias[col];
      }
    }
  }
}

// ---------------------------------------------------------------------------
extern "C" void kernel_launch(void* const* d_in, const int* in_sizes, int n_in,
                              void* d_out, int out_size, void* d_ws, size_t ws_size,
                              hipStream_t stream) {
  const float* x     = (const float*)d_in[0];
  const float* w_qkv = (const float*)d_in[1];
  const float* b_qkv = (const float*)d_in[2];
  const float* w_out = (const float*)d_in[3];
  const float* b_out = (const float*)d_in[4];

  bf16* qkv_ws  = (bf16*)d_ws;                               // 3*4096*1024 bf16
  bf16* attn_ws = qkv_ws + (size_t)3 * Mrows * Cc;           // 4096*1024 bf16

  ta_qkv_gemm<<<dim3(N3 / 128, Mrows / 128), 256, 0, stream>>>(
      x, w_qkv, b_qkv, qkv_ws);
  ta_flash_attn<<<dim3(Nseq / 128, Bb * Hh), 256, 0, stream>>>(
      qkv_ws, attn_ws);
  ta_out_gemm<<<dim3(Cc / 128, Mrows / 128), 256, 0, stream>>>(
      attn_ws, w_out, b_out, (float*)d_out);
}
